// CatNet_14809047236966
// MI455X (gfx1250) — matrix operand order repaired and checked by reading, not compiled, not run on hardware
//
#include <hip/hip_runtime.h>

// CDNA5 / gfx1250 fused SNN conv stack, round 4:
//   - f16 activations (spikes exactly {0,1}), f32 WMMA accumulate
//   - implicit-GEMM conv via v_wmma_f32_16x16x32_f16
//   - wave owns one pixel x CGW channel groups (A fragments reused)
//   - patch staging via TENSOR_LOAD_TO_LDS (TDM, 6-arg clang-23 builtin) with
//     double-buffered LDS patch -> DMA of ci-tile cb+1 overlaps WMMA of tile cb.
//     Border blocks / padded layers fall back to manual staging.
//   - integrate-and-fire scan fused in-kernel through LDS

typedef __attribute__((ext_vector_type(16))) _Float16     v16h;
typedef __attribute__((ext_vector_type(8)))  float        v8f;
typedef __attribute__((ext_vector_type(4)))  float        v4f;
typedef __attribute__((ext_vector_type(4)))  unsigned int v4u;

typedef __attribute__((ext_vector_type(4))) unsigned int tdm_u4;
typedef __attribute__((ext_vector_type(8))) int          tdm_i8;
typedef __attribute__((ext_vector_type(4))) int          tdm_i4;

// ---------------------------------------------------------------- prep kernels

__global__ void cvt_f32_to_f16(const float* __restrict__ s, _Float16* __restrict__ d, int n) {
  int i = blockIdx.x * blockDim.x + threadIdx.x;
  if (i < n) d[i] = (_Float16)s[i];
}

// weights (O, CI, KH, KW) f32 -> f16, re-blocked as [o][cb][KLOCpad] with a
// zero tail per ci-tile block (cb), so each 32-wide K chunk is self-contained.
__global__ void cvt_w_pad(const float* __restrict__ w, _Float16* __restrict__ wp,
                          int O, int ktot, int cib, int kloc, int klp) {
  int i = blockIdx.x * blockDim.x + threadIdx.x;
  if (i >= O * cib * klp) return;
  int o  = i / (cib * klp);
  int rm = i % (cib * klp);
  int cb = rm / klp;
  int r  = rm % klp;
  wp[i] = (r < kloc) ? (_Float16)w[o * ktot + cb * kloc + r] : (_Float16)0.0f;
}

// ---------------------------------------------------------------- TDM helper

// 3D tile load: dim0 = t0 contiguous elems (2B), dim1 = t1 rows (stride s0),
// dim2 = t2 planes (stride s1). Packed LDS write order == [z][y][x].
static __device__ __forceinline__ void tdm_load_3d(unsigned lds_addr,
                                                   unsigned long long gaddr,
                                                   unsigned t0, unsigned t1, unsigned t2,
                                                   unsigned long long s0,
                                                   unsigned long long s1) {
  tdm_u4 g0;
  g0[0] = 1u;                                             // count=1, user mode
  g0[1] = lds_addr;                                       // [63:32] lds_addr
  g0[2] = (unsigned)(gaddr & 0xffffffffull);              // [95:64] global_addr lo
  g0[3] = (unsigned)((gaddr >> 32) & 0x01ffffffull)       // [120:96] global_addr hi
        | (2u << 30);                                     // [127:126] type=2 (image)
  tdm_i8 g1;
  g1[0] = (int)(1u << 16);                                // data_size=1 (2 bytes)
  g1[1] = (int)((t0 & 0xffffu) << 16);                    // tensor_dim0 lo16
  g1[2] = (int)(((t0 >> 16) & 0xffffu) | ((t1 & 0xffffu) << 16)); // dim0 hi | dim1 lo
  g1[3] = (int)(((t1 >> 16) & 0xffffu) | ((t0 & 0xffffu) << 16)); // dim1 hi | tile_dim0
  g1[4] = (int)((t1 & 0xffffu) | ((t2 & 0xffffu) << 16)); // tile_dim1 | tile_dim2
  g1[5] = (int)(s0 & 0xffffffffull);                      // stride0 lo32
  g1[6] = (int)(((s0 >> 32) & 0xffffull) | ((s1 & 0xffffull) << 16)); // s0 hi | s1 lo16
  g1[7] = (int)((s1 >> 16) & 0xffffffffull);              // stride1 [47:16]
  tdm_i4 g2;
  g2[0] = (int)t2;                                        // tensor_dim2
  g2[1] = 0; g2[2] = 0; g2[3] = 0;                        // dim3/stride2/tile_dim3 = 0
  tdm_i4 g3 = {0, 0, 0, 0};
  tdm_i8 g4 = {0, 0, 0, 0, 0, 0, 0, 0};                   // extra group (clang-23 form)
  __builtin_amdgcn_tensor_load_to_lds(g0, g1, g2, g3, g4, 0);
}

// ---------------------------------------------------------------- conv + spike

// compile-time LDS offset (in halfs) of local im2col k within the staged patch
template<int KS, int PW, int KLOC>
constexpr int lds_off(int k) {
  return (k >= KLOC) ? 0
       : (((k / (KS * KS)) * KS + (k % (KS * KS)) / KS) * PW + (k % KS)) * 32;
}

template<int CIN, int COUT, int KS, int STR, int PADV,
         int HIN, int WIN, int HOUT, int WOUT,
         int CITILE, int PPB, int CGW, int DBUF>
__global__ __launch_bounds__(256)
void conv_spike(const _Float16* __restrict__ in, const _Float16* __restrict__ wt,
                const float* __restrict__ bias, _Float16* __restrict__ out) {
  constexpr int T    = 32;
  constexpr int CG   = COUT / 16;                 // 16-channel groups
  static_assert(PPB * (CG / CGW) == 8, "wave budget must be 8 waves/block");
  constexpr int PW   = (PPB - 1) * STR + KS;      // staged patch width
  constexpr int XT   = (WOUT + PPB - 1) / PPB;
  constexpr int KLOC = CITILE * KS * KS;          // K per ci-tile (unpadded)
  constexpr int KLP  = ((KLOC + 31) / 32) * 32;   // padded to chunk multiple
  constexpr int KCHB = KLP / 32;
  constexpr int CIB  = CIN / CITILE;
  constexpr int KPAD = CIB * KLP;                 // padded weight row length
  constexpr int PSZ  = CITILE * KS * PW * T;      // patch elems per buffer

  __shared__ __align__(16) _Float16 patch[(DBUF ? 2 : 1) * PSZ];
  __shared__ __align__(16) float    sbuf[8][16][T];               // 16 KB

  const int bid = blockIdx.x;
  const int n   = bid / (HOUT * XT);
  const int rm  = bid % (HOUT * XT);
  const int oy  = rm / XT;
  const int x0  = (rm % XT) * PPB;

  const int tid  = threadIdx.x;
  const int wave = tid >> 5;
  const int lane = tid & 31;
  const int mrow = lane & 15;
  const int hi   = lane >> 4;

  const int pixL  = wave % PPB;                   // pixel owned by this wave
  const int cgrp0 = (wave / PPB) * CGW;           // first channel group
  const int ox    = x0 + pixL;
  const int pxo   = pixL * STR;                   // patch column of kx=0

  const int iy0 = oy * STR - PADV;
  const int ix0 = x0 * STR - PADV;

  // receptive field fully in-bounds -> TDM path (uniform per block)
  bool useTDM = false;
  if constexpr (DBUF) {
    useTDM = (iy0 >= 0) && (ix0 >= 0) && (iy0 + KS <= HIN) && (ix0 + PW <= WIN);
  }

  v8f acc0[CGW];   // rows t = 0..15 per channel group
  v8f acc1[CGW];   // rows t = 16..31
#pragma unroll
  for (int g = 0; g < CGW; ++g) { acc0[g] = {}; acc1[g] = {}; }

  // issue TDM for ci-tile 0 before the loop (wave 0 only)
  if constexpr (DBUF) {
    if (useTDM && wave == 0) {
      const long goff = (((long)(n * CIN) * HIN + iy0) * WIN + ix0) * T;
      tdm_load_3d((unsigned)(unsigned long long)(void*)&patch[0],
                  (unsigned long long)(const void*)(in + goff),
                  (unsigned)(PW * T), (unsigned)KS, (unsigned)CITILE,
                  (unsigned long long)(WIN * T),
                  (unsigned long long)(HIN * WIN * T));
    }
  }

  for (int cb = 0; cb < CIB; ++cb) {
    _Float16* pb = &patch[(DBUF ? (cb & 1) : 0) * PSZ];

    if (DBUF && useTDM) {
      if constexpr (DBUF) {
        if (wave == 0) {
          if (cb + 1 < CIB) {   // prefetch next ci-tile into the other buffer
            const long goff =
                (((long)(n * CIN + (cb + 1) * CITILE) * HIN + iy0) * WIN + ix0) * T;
            tdm_load_3d((unsigned)(unsigned long long)(void*)&patch[((cb + 1) & 1) * PSZ],
                        (unsigned long long)(const void*)(in + goff),
                        (unsigned)(PW * T), (unsigned)KS, (unsigned)CITILE,
                        (unsigned long long)(WIN * T),
                        (unsigned long long)(HIN * WIN * T));
            __builtin_amdgcn_s_wait_tensorcnt(1);   // tile cb complete
          } else {
            __builtin_amdgcn_s_wait_tensorcnt(0);
          }
        }
      }
    } else {
      // ---- manual staging (border blocks / padded layers) ----
      constexpr int ITEMS = CITILE * KS * PW;
      for (int it = tid; it < ITEMS; it += 256) {
        const int ci = cb * CITILE + it / (KS * PW);
        const int r  = it % (KS * PW);
        const int iy = iy0 + r / PW;
        const int ix = ix0 + r % PW;
        v4u* d4 = (v4u*)&pb[it * T];
        if (iy >= 0 && iy < HIN && ix >= 0 && ix < WIN) {
          const v4u* s4 = (const v4u*)&in[((((long)n * CIN + ci) * HIN + iy) * WIN + ix) * T];
          d4[0] = s4[0]; d4[1] = s4[1]; d4[2] = s4[2]; d4[3] = s4[3];
        } else {
          v4u z = {0u, 0u, 0u, 0u};
          d4[0] = z; d4[1] = z; d4[2] = z; d4[3] = z;
        }
      }
    }
    __syncthreads();   // patch for tile cb visible to all waves

    const long wbase = (long)cb * KLP;
#pragma unroll
    for (int c = 0; c < KCHB; ++c) {
      // ---- A fragments (gathered ONCE, reused for CGW WMMAs) ----
      v16h a0 = {}, a1 = {};
      const int abase = pxo * T + mrow;
#pragma unroll
      for (int e = 0; e < 16; ++e) {
        const int kb  = c * 32 + e;                          // lane<16 K index
        const int o0  = lds_off<KS, PW, KLOC>(kb);          // compile-time
        const int o1  = lds_off<KS, PW, KLOC>(kb + 16);     // lane>=16 K index
        const int off = (hi ? o1 : o0) + abase;
        a0[e] = pb[off];        // t = mrow       (tile 0)
        a1[e] = pb[off + 16];   // t = mrow + 16  (tile 1)
      }
      // ---- CGW channel groups share the A fragments ----
#pragma unroll
      for (int g = 0; g < CGW; ++g) {
        v16h bfrag;
        {
          const _Float16* bp =
              &wt[(long)((cgrp0 + g) * 16 + mrow) * KPAD + wbase + c * 32 + 16 * hi];
          union { v4u u[2]; v16h h; } ub;
          ub.u[0] = ((const v4u*)bp)[0];
          ub.u[1] = ((const v4u*)bp)[1];
          bfrag = ub.h;
        }
        acc0[g] = __builtin_amdgcn_wmma_f32_16x16x32_f16(false, a0, false, bfrag,
                                                         (short)0, acc0[g], false, false);
        acc1[g] = __builtin_amdgcn_wmma_f32_16x16x32_f16(false, a1, false, bfrag,
                                                         (short)0, acc1[g], false, false);
      }
    }
    __syncthreads();   // all waves done reading buffer cb before it is reused
  }

  // ---- spike scan per channel group (LDS transpose of C tiles) ----
#pragma unroll
  for (int g = 0; g < CGW; ++g) {
    __syncthreads();
    {
      float* sp = &sbuf[wave][mrow][hi * 8];
      v4f t0a = {acc0[g][0], acc0[g][1], acc0[g][2], acc0[g][3]};
      v4f t0b = {acc0[g][4], acc0[g][5], acc0[g][6], acc0[g][7]};
      v4f t1a = {acc1[g][0], acc1[g][1], acc1[g][2], acc1[g][3]};
      v4f t1b = {acc1[g][4], acc1[g][5], acc1[g][6], acc1[g][7]};
      *(v4f*)&sp[0]  = t0a; *(v4f*)&sp[4]  = t0b;   // t = 8*hi + 0..7
      *(v4f*)&sp[16] = t1a; *(v4f*)&sp[20] = t1b;   // t = 16 + 8*hi + 0..7
    }
    __syncthreads();

    if (lane < 16 && ox < WOUT) {
      const int och = (cgrp0 + g) * 16 + lane;
      const float b = bias[och];
      union { v4f v[8]; float f[32]; } rw;
      const v4f* rp = (const v4f*)&sbuf[wave][lane][0];
#pragma unroll
      for (int q = 0; q < 8; ++q) rw.v[q] = rp[q];
      union { _Float16 h[32]; v4u u[4]; } sp;
      float v = 0.0f;
#pragma unroll
      for (int t = 0; t < T; ++t) {
        v += rw.f[t] + b;
        const float s = (v >= 1.0f) ? 1.0f : 0.0f;
        v -= s;
        sp.h[t] = (_Float16)s;
      }
      _Float16* op = &out[(((long)(n * COUT + och) * HOUT + oy) * WOUT + ox) * T];
      v4u* o4 = (v4u*)op;
      o4[0] = sp.u[0]; o4[1] = sp.u[1]; o4[2] = sp.u[2]; o4[3] = sp.u[3];
    }
  }
}

// ---------------------------------------------------------------- FC head
// out[n,o] = bfc[o] + sum_j wfc[o,j] * (sum_t h7[n,j,t]) / 32   (time-sum commutes)

__global__ void fc_reduce(const _Float16* __restrict__ h7, float* __restrict__ hbar) {
  int i = blockIdx.x * blockDim.x + threadIdx.x;      // 128*512
  if (i >= 128 * 512) return;
  const _Float16* p = &h7[(long)i * 32];
  float s = 0.0f;
#pragma unroll
  for (int t = 0; t < 32; ++t) s += (float)p[t];
  hbar[i] = s * (1.0f / 32.0f);
}

__global__ void fc_out_k(const float* __restrict__ hbar, const float* __restrict__ wfc,
                         const float* __restrict__ bfc, float* __restrict__ outp) {
  int i = blockIdx.x * blockDim.x + threadIdx.x;      // 128*10
  if (i >= 128 * 10) return;
  int n = i / 10, o = i % 10;
  float s = bfc[o];
  const float* hb = &hbar[n * 512];
  const float* wp = &wfc[o * 512];
  for (int j = 0; j < 512; ++j) s += hb[j] * wp[j];
  outp[i] = s;
}

// ---------------------------------------------------------------- launch

extern "C" void kernel_launch(void* const* d_in, const int* in_sizes, int n_in,
                              void* d_out, int out_size, void* d_ws, size_t ws_size,
                              hipStream_t stream) {
  (void)in_sizes; (void)n_in; (void)out_size; (void)ws_size;
  const float* x   = (const float*)d_in[0];
  const float* w1  = (const float*)d_in[1];  const float* b1 = (const float*)d_in[2];
  const float* w2  = (const float*)d_in[3];  const float* b2 = (const float*)d_in[4];
  const float* w3  = (const float*)d_in[5];  const float* b3 = (const float*)d_in[6];
  const float* w4  = (const float*)d_in[7];  const float* b4 = (const float*)d_in[8];
  const float* w5  = (const float*)d_in[9];  const float* b5 = (const float*)d_in[10];
  const float* w6  = (const float*)d_in[11]; const float* b6 = (const float*)d_in[12];
  const float* w7  = (const float*)d_in[13]; const float* b7 = (const float*)d_in[14];
  const float* wfc = (const float*)d_in[15]; const float* bfc = (const float*)d_in[16];

  char* ws = (char*)d_ws;
  size_t off = 0;
  auto take = [&](size_t bytes) -> char* {
    char* p = ws + off;
    off = (off + bytes + 255) & ~(size_t)255;
    return p;
  };
  _Float16* xh  = (_Float16*)take(3211264ull * 2);    // x as f16, T innermost
  _Float16* P0  = (_Float16*)take(88604672ull * 2);   // max(h1,h3,h5,h7)
  _Float16* P1  = (_Float16*)take(75497472ull * 2);   // max(h2,h4,h6)
  _Float16* wp1 = (_Float16*)take((size_t)32  * 32   * 2);   // CIB*KLP = 1*32
  _Float16* wp2 = (_Float16*)take((size_t)32  * 320  * 2);   // 2*160
  _Float16* wp3 = (_Float16*)take((size_t)32  * 512  * 2);   // 4*128
  _Float16* wp4 = (_Float16*)take((size_t)64  * 320  * 2);   // 2*160
  _Float16* wp5 = (_Float16*)take((size_t)64  * 640  * 2);   // 4*160
  _Float16* wp6 = (_Float16*)take((size_t)64  * 1024 * 2);   // 8*128
  _Float16* wp7 = (_Float16*)take((size_t)128 * 576  * 2);   // 2*288
  float*   hbar = (float*)take(65536ull * 4);

  // precision conversion / per-ci-tile weight padding
  cvt_f32_to_f16<<<(3211264 + 255) / 256, 256, 0, stream>>>(x, xh, 3211264);
  cvt_w_pad<<<(32 * 32    + 255) / 256, 256, 0, stream>>>(w1, wp1, 32,  9,    1, 9,   32);
  cvt_w_pad<<<(32 * 320   + 255) / 256, 256, 0, stream>>>(w2, wp2, 32,  288,  2, 144, 160);
  cvt_w_pad<<<(32 * 512   + 255) / 256, 256, 0, stream>>>(w3, wp3, 32,  512,  4, 128, 128);
  cvt_w_pad<<<(64 * 320   + 255) / 256, 256, 0, stream>>>(w4, wp4, 64,  288,  2, 144, 160);
  cvt_w_pad<<<(64 * 640   + 255) / 256, 256, 0, stream>>>(w5, wp5, 64,  576,  4, 144, 160);
  cvt_w_pad<<<(64 * 1024  + 255) / 256, 256, 0, stream>>>(w6, wp6, 64,  1024, 8, 128, 128);
  cvt_w_pad<<<(128 * 576  + 255) / 256, 256, 0, stream>>>(w7, wp7, 128, 576,  2, 288, 288);

  // fused conv+spike chain; grid = N * HOUT * ceil(WOUT/PPB)
  // template args: CIN,COUT,KS,STR,PAD,HIN,WIN,HOUT,WOUT,CITILE,PPB,CGW,DBUF
  conv_spike<1,  32, 3, 1, 0, 28, 28, 26, 26, 1,  8, 2, 1><<<128 * 26 * 4, 256, 0, stream>>>(xh, wp1, b1, P0);
  conv_spike<32, 32, 3, 1, 0, 26, 26, 24, 24, 16, 8, 2, 1><<<128 * 24 * 3, 256, 0, stream>>>(P0, wp2, b2, P1);
  conv_spike<32, 32, 4, 2, 1, 24, 24, 12, 12, 8,  8, 2, 0><<<128 * 12 * 2, 256, 0, stream>>>(P1, wp3, b3, P0);
  conv_spike<32, 64, 3, 1, 0, 12, 12, 10, 10, 16, 8, 4, 1><<<128 * 10 * 2, 256, 0, stream>>>(P0, wp4, b4, P1);
  conv_spike<64, 64, 3, 1, 0, 10, 10, 8,  8,  16, 8, 4, 1><<<128 * 8  * 1, 256, 0, stream>>>(P1, wp5, b5, P0);
  conv_spike<64, 64, 4, 2, 1, 8,  8,  4,  4,  8,  4, 2, 0><<<128 * 4  * 1, 256, 0, stream>>>(P0, wp6, b6, P1);
  conv_spike<64, 128,3, 1, 0, 4,  4,  2,  2,  32, 2, 2, 1><<<128 * 2  * 1, 256, 0, stream>>>(P1, wp7, b7, P0);

  // FC head
  fc_reduce<<<65536 / 256, 256, 0, stream>>>(P0, hbar);
  fc_out_k<<<(1280 + 255) / 256, 256, 0, stream>>>(hbar, wfc, bfc, (float*)d_out);
}
